// Decoder_12386685682081
// MI455X (gfx1250) — compile-verified, matrix operand order checked by reading, not run on hardware
//
#include <hip/hip_runtime.h>
#include <hip/hip_bf16.h>
#include <math.h>

// Problem constants (match reference)
#define B_  32
#define T_  128
#define S_  256
#define E_  512
#define H_  1024
#define O_  32000
#define SOS 2

typedef __bf16  v8bf  __attribute__((ext_vector_type(8)));
typedef __bf16  v16bf __attribute__((ext_vector_type(16)));
typedef float   v8f   __attribute__((ext_vector_type(8)));

// ---------------------------------------------------------------------------
// fp32 -> bf16 weight conversion (one-time per launch)
// ---------------------------------------------------------------------------
__global__ __launch_bounds__(256) void cvt_f32_bf16(const float* __restrict__ src,
                                                    __bf16* __restrict__ dst,
                                                    long n) {
    long i = (long)blockIdx.x * 1024 + threadIdx.x;
#pragma unroll
    for (int k = 0; k < 4; ++k, i += 256)
        if (i < n) dst[i] = (__bf16)src[i];
}

// ---------------------------------------------------------------------------
// init: h = encoder_h_t ; hx = [bf16(h) | 0] ; y = SOS
// ---------------------------------------------------------------------------
__global__ __launch_bounds__(256) void init_state(const float* __restrict__ h0,
                                                  float* __restrict__ h,
                                                  __bf16* __restrict__ hx,
                                                  int* __restrict__ y) {
    const int b = blockIdx.x, tid = threadIdx.x;
    for (int j = tid; j < H_; j += 256) {
        float v = h0[b * H_ + j];
        h[b * H_ + j] = v;
        hx[(size_t)b * 2 * H_ + j] = (__bf16)v;
        hx[(size_t)b * 2 * H_ + H_ + j] = (__bf16)0.0f;
    }
    if (tid == 0) y[b] = SOS;
}

// ---------------------------------------------------------------------------
// x_bf16 = [ bf16(emb[y]) | bf16(ctx) ]   (ctx already bf16 in hx hi-half)
// ---------------------------------------------------------------------------
__global__ __launch_bounds__(256) void build_x(const int* __restrict__ y,
                                               const float* __restrict__ emb,
                                               const __bf16* __restrict__ hx,
                                               __bf16* __restrict__ x) {
    const int b = blockIdx.x, tid = threadIdx.x;
    const int yy = y[b];
    for (int j = tid; j < E_; j += 256)
        x[(size_t)b * (E_ + H_) + j] = (__bf16)emb[(size_t)yy * E_ + j];
    for (int j = tid; j < H_; j += 256)
        x[(size_t)b * (E_ + H_) + E_ + j] = hx[(size_t)b * 2 * H_ + H_ + j];
}

// ---------------------------------------------------------------------------
// WMMA bf16 GEMM:  D[m][n] = sum_k A[m][k] * Bw[n][k]  (+ bias[n])
//   A : [32, K]  row stride lda (bf16)        (activations)
//   Bw: [N, K]  row stride K   (bf16)        (weights, as given: [N,K])
//   D : fp32, row stride ldd
// One 256-thread block (8 wave32 waves) computes a 32 x 64 output tile:
//   wave w: m-tile = w&1, n-tile = w>>2? no: n-tile = w>>1 -> 2x4 tiles of 16x16.
// A/B fragment layout per CDNA5 ISA (16-bit 16x32 A):
//   lane l: row/col = l&15, kbase = (l>>4)*8, elems j<8 -> k=kbase+j,
//   j>=8 -> k=kbase+8+j  (i.e. two contiguous 8-element (16B) segments).
// ---------------------------------------------------------------------------
__global__ __launch_bounds__(256) void gemm_bf16_wmma(const __bf16* __restrict__ A, int lda,
                                                      const __bf16* __restrict__ Bw,
                                                      const float* __restrict__ bias,
                                                      float* __restrict__ D, size_t ldd,
                                                      int K) {
    const int tid  = threadIdx.x;
    const int wave = tid >> 5;
    const int lane = tid & 31;
    const int mt   = wave & 1;       // 0/1 : rows 0-15 / 16-31
    const int nt   = wave >> 1;      // 0..3: 16-col slabs
    const int half = lane >> 4;      // 0/1
    const int l16  = lane & 15;
    const int kb   = half << 3;      // 0 or 8

    const int     ncol = blockIdx.x * 64 + nt * 16 + l16;
    const __bf16* arow = A  + (size_t)(mt * 16 + l16) * lda;
    const __bf16* brow = Bw + (size_t)ncol * K;

    v8f acc = {0.f, 0.f, 0.f, 0.f, 0.f, 0.f, 0.f, 0.f};

#pragma unroll 4
    for (int k0 = 0; k0 < K; k0 += 32) {
        v8bf a0 = *(const v8bf*)(arow + k0 + kb);
        v8bf a1 = *(const v8bf*)(arow + k0 + kb + 16);
        v8bf b0 = *(const v8bf*)(brow + k0 + kb);
        v8bf b1 = *(const v8bf*)(brow + k0 + kb + 16);
        __builtin_prefetch(brow + k0 + 256, 0, 1);   // global_prefetch_b8 down the weight stream
        v16bf av = __builtin_shufflevector(a0, a1, 0,1,2,3,4,5,6,7,8,9,10,11,12,13,14,15);
        v16bf bv = __builtin_shufflevector(b0, b1, 0,1,2,3,4,5,6,7,8,9,10,11,12,13,14,15);
        acc = __builtin_amdgcn_wmma_f32_16x16x32_bf16(false, av, false, bv,
                                                      (short)0, acc, false, false);
    }

    const float bval = bias ? bias[ncol] : 0.0f;
    // C/D layout: VGPR v holds M = v + 8*half (within 16-row tile), N = lane&15
#pragma unroll
    for (int v = 0; v < 8; ++v) {
        int m = mt * 16 + half * 8 + v;
        D[(size_t)m * ldd + ncol] = acc[v] + bval;
    }
}

// ---------------------------------------------------------------------------
// GRU cell update (elementwise over [32,1024])
// ---------------------------------------------------------------------------
__global__ __launch_bounds__(256) void gru_update(const float* __restrict__ gi,
                                                  const float* __restrict__ gh,
                                                  float* __restrict__ h,
                                                  __bf16* __restrict__ hx) {
    const int b = blockIdx.x, tid = threadIdx.x;
    for (int j = tid; j < H_; j += 256) {
        const size_t gb = (size_t)b * 3 * H_;
        float ir = gi[gb + j],          hr = gh[gb + j];
        float iz = gi[gb + H_ + j],     hz = gh[gb + H_ + j];
        float in = gi[gb + 2 * H_ + j], hn = gh[gb + 2 * H_ + j];
        float r = 1.0f / (1.0f + expf(-(ir + hr)));
        float z = 1.0f / (1.0f + expf(-(iz + hz)));
        float n = tanhf(in + r * hn);
        float hv = (1.0f - z) * n + z * h[(size_t)b * H_ + j];
        h[(size_t)b * H_ + j] = hv;
        hx[(size_t)b * 2 * H_ + j] = (__bf16)hv;
    }
}

// ---------------------------------------------------------------------------
// Fused masked attention: scores -> softmax -> ctx ; ctx written bf16 into
// the hi half of hx. One block per batch row.
// ---------------------------------------------------------------------------
__global__ __launch_bounds__(256) void attention(const float* __restrict__ eo,
                                                 const unsigned char* __restrict__ mask,
                                                 const float* __restrict__ h,
                                                 __bf16* __restrict__ hx) {
    __shared__ float hvec[H_];
    __shared__ float sc[S_];
    __shared__ float red[S_];
    const int b = blockIdx.x, tid = threadIdx.x;

    for (int j = tid; j < H_; j += 256) hvec[j] = h[(size_t)b * H_ + j];
    __syncthreads();

    // one score per thread (S_ == 256)
    const float* row = eo + ((size_t)b * S_ + tid) * H_;
    float acc = 0.0f;
#pragma unroll 4
    for (int k = 0; k < H_; k += 4)
        acc += row[k] * hvec[k] + row[k + 1] * hvec[k + 1] +
               row[k + 2] * hvec[k + 2] + row[k + 3] * hvec[k + 3];
    if (!mask[(size_t)b * S_ + tid]) acc = -1e9f;
    sc[tid] = acc; red[tid] = acc;
    __syncthreads();
    for (int off = 128; off > 0; off >>= 1) {
        if (tid < off) red[tid] = fmaxf(red[tid], red[tid + off]);
        __syncthreads();
    }
    const float mx = red[0];
    __syncthreads();
    float e = expf(sc[tid] - mx);
    sc[tid] = e; red[tid] = e;
    __syncthreads();
    for (int off = 128; off > 0; off >>= 1) {
        if (tid < off) red[tid] += red[tid + off];
        __syncthreads();
    }
    const float inv = 1.0f / red[0];
    __syncthreads();
    sc[tid] *= inv;
    __syncthreads();

    // ctx[b][j] = sum_s p[s] * eo[b][s][j]
    for (int j = tid; j < H_; j += 256) {
        const float* col = eo + (size_t)b * S_ * H_ + j;
        float a = 0.0f;
#pragma unroll 4
        for (int s = 0; s < S_; ++s) a += sc[s] * col[(size_t)s * H_];
        hx[(size_t)b * 2 * H_ + H_ + j] = (__bf16)a;
    }
}

// ---------------------------------------------------------------------------
// argmax over O_ logits, jnp semantics (first index on ties)
// P points at out + t*O_, row stride ldd = T_*O_
// ---------------------------------------------------------------------------
__global__ __launch_bounds__(256) void argmax_k(const float* __restrict__ P, size_t ldd,
                                                int* __restrict__ y) {
    __shared__ float sv[256];
    __shared__ int   si[256];
    const int b = blockIdx.x, tid = threadIdx.x;
    float best = -3.4e38f; int bi = O_;
    for (int o = tid; o < O_; o += 256) {
        float v = P[(size_t)b * ldd + o];
        if (v > best) { best = v; bi = o; }          // strict > keeps first per-thread
    }
    sv[tid] = best; si[tid] = bi;
    __syncthreads();
    for (int off = 128; off > 0; off >>= 1) {
        if (tid < off) {
            if (sv[tid + off] > sv[tid] ||
                (sv[tid + off] == sv[tid] && si[tid + off] < si[tid])) {
                sv[tid] = sv[tid + off]; si[tid] = si[tid + off];
            }
        }
        __syncthreads();
    }
    if (tid == 0) y[b] = si[0];
}

// ---------------------------------------------------------------------------
extern "C" void kernel_launch(void* const* d_in, const int* in_sizes, int n_in,
                              void* d_out, int out_size, void* d_ws, size_t ws_size,
                              hipStream_t stream) {
    (void)in_sizes; (void)n_in; (void)out_size; (void)ws_size;
    // inputs per setup_inputs() order
    const float*         eo   = (const float*)d_in[1];          // [B,S,H]
    const float*         h0   = (const float*)d_in[2];          // [B,H]
    const unsigned char* mask = (const unsigned char*)d_in[3];  // [B,S] bool
    const float*         emb  = (const float*)d_in[4];          // [V,E]
    const float*         W_ih = (const float*)d_in[5];          // [3H, E+H]
    const float*         W_hh = (const float*)d_in[6];          // [3H, H]
    const float*         b_ih = (const float*)d_in[7];
    const float*         b_hh = (const float*)d_in[8];
    const float*         W_c  = (const float*)d_in[9];          // [O, 2H]
    const float*         b_c  = (const float*)d_in[10];
    float* out = (float*)d_out;                                 // [B,T,O]

    // workspace carve-up
    char* ws = (char*)d_ws;
    size_t off = 0;
    auto take = [&](size_t bytes) {
        char* p = ws + off;
        off = (off + bytes + 255) & ~(size_t)255;
        return p;
    };
    __bf16* Wih_bf = (__bf16*)take((size_t)3 * H_ * (E_ + H_) * sizeof(__bf16));
    __bf16* Whh_bf = (__bf16*)take((size_t)3 * H_ * H_ * sizeof(__bf16));
    __bf16* Wc_bf  = (__bf16*)take((size_t)O_ * 2 * H_ * sizeof(__bf16));
    float*  h_st   = (float*)take((size_t)B_ * H_ * sizeof(float));
    float*  gi     = (float*)take((size_t)B_ * 3 * H_ * sizeof(float));
    float*  gh     = (float*)take((size_t)B_ * 3 * H_ * sizeof(float));
    __bf16* hx_bf  = (__bf16*)take((size_t)B_ * 2 * H_ * sizeof(__bf16)); // [h|ctx]
    __bf16* x_bf   = (__bf16*)take((size_t)B_ * (E_ + H_) * sizeof(__bf16));
    int*    y_st   = (int*)take((size_t)B_ * sizeof(int));

    // --- prologue: bf16 weight images + initial state ---
    {
        long n1 = (long)3 * H_ * (E_ + H_);
        long n2 = (long)3 * H_ * H_;
        long n3 = (long)O_ * 2 * H_;
        cvt_f32_bf16<<<(unsigned)((n1 + 1023) / 1024), 256, 0, stream>>>(W_ih, Wih_bf, n1);
        cvt_f32_bf16<<<(unsigned)((n2 + 1023) / 1024), 256, 0, stream>>>(W_hh, Whh_bf, n2);
        cvt_f32_bf16<<<(unsigned)((n3 + 1023) / 1024), 256, 0, stream>>>(W_c,  Wc_bf,  n3);
        init_state<<<B_, 256, 0, stream>>>(h0, h_st, hx_bf, y_st);
    }

    const size_t ldOut = (size_t)T_ * O_;

    for (int t = 0; t < T_; ++t) {
        // x = [emb[y] | ctx]
        build_x<<<B_, 256, 0, stream>>>(y_st, emb, hx_bf, x_bf);
        // gi = x @ W_ih^T + b_ih    (K=1536, N=3072)
        gemm_bf16_wmma<<<3 * H_ / 64, 256, 0, stream>>>(x_bf, E_ + H_, Wih_bf, b_ih,
                                                        gi, (size_t)(3 * H_), E_ + H_);
        // gh = h @ W_hh^T + b_hh    (K=1024, N=3072)
        gemm_bf16_wmma<<<3 * H_ / 64, 256, 0, stream>>>(hx_bf, 2 * H_, Whh_bf, b_hh,
                                                        gh, (size_t)(3 * H_), H_);
        // GRU update -> h, hx[:,0:1024]
        gru_update<<<B_, 256, 0, stream>>>(gi, gh, h_st, hx_bf);
        // attention -> hx[:,1024:2048]
        attention<<<B_, 256, 0, stream>>>(eo, mask, h_st, hx_bf);
        // pred = [h|ctx] @ W_c^T + b_c -> out[:, t, :]   (K=2048, N=32000)
        gemm_bf16_wmma<<<O_ / 64, 256, 0, stream>>>(hx_bf, 2 * H_, Wc_bf, b_c,
                                                    out + (size_t)t * O_, ldOut, 2 * H_);
        // y = argmax(pred)
        argmax_k<<<B_, 256, 0, stream>>>(out + (size_t)t * O_, ldOut, y_st);
    }
}
